// AgentAttention_60782377173176
// MI455X (gfx1250) — matrix operand order repaired and checked by reading, not compiled, not run on hardware
//
#include <hip/hip_runtime.h>
#include <stdint.h>

// ---------------- problem constants (fixed by setup_inputs) ----------------
#define BATCH   32
#define NPIX    3136          // 56*56
#define CDIM    384
#define NHEADS  8
#define ANUM    16
#define DHEAD   48
#define POOLB   14            // 56/4
#define SCALE   0.14433756729740643f   // 48^-0.5
#define ROWTILE 64
#define NTILES  (NPIX / ROWTILE)       // 49
#define TILE_ELEMS (ROWTILE * CDIM)    // 24576 f32 elements, contiguous in memory

typedef __attribute__((ext_vector_type(16))) __bf16   v16bf;
typedef __attribute__((ext_vector_type(8)))  float    v8f;
typedef __attribute__((ext_vector_type(4)))  unsigned u32x4;
typedef __attribute__((ext_vector_type(8)))  int      i32x8;
typedef __attribute__((ext_vector_type(4)))  int      i32x4;
typedef __attribute__((ext_vector_type(4)))  float    f32x4;
typedef __attribute__((ext_vector_type(16))) float    f32x16;

union Frag  { v16bf v; u32x4 u[2]; };
union FragF { f32x16 f; f32x4 q[4]; };

__device__ __forceinline__ unsigned short f2bf(float f) {
  union { float f; unsigned u; } c; c.f = f;
  unsigned u = c.u;
  u += 0x7fffu + ((u >> 16) & 1u);      // round-to-nearest-even truncation
  return (unsigned short)(u >> 16);
}

// lane-xor reduction helper: ds_swizzle group-of-32, pattern (xor<<10)|0x1f
#define SWZ_F(v, imm) \
  __int_as_float(__builtin_amdgcn_ds_swizzle(__float_as_int(v), (imm)))

// A fragment (bf16 source, row-major LDS). 16-bit A 16x32 layout:
// lanes 0-15 row M=l, K 0..7 & 16..23; lanes 16-31 same rows, K 8..15 & 24..31.
__device__ __forceinline__ v16bf load_a_bf(const unsigned short* m, int r0, int kbase,
                                           int ld, int lane) {
  int l16 = lane & 15, half = lane >> 4;
  const unsigned short* p = m + (size_t)(r0 + l16) * ld + kbase + half * 8;
  Frag f;
  f.u[0] = *(const u32x4*)(p);        // K +0..7   (pairs per dword)
  f.u[1] = *(const u32x4*)(p + 16);   // K +16..23
  return f.v;
}

// A fragment built from f32 LDS tile (TDM-copied raw) with on-the-fly bf16 cvt.
__device__ __forceinline__ v16bf load_a_f32(const float* m, int r0, int kbase,
                                            int ld, int lane) {
  int l16 = lane & 15, half = lane >> 4;
  const float* p = m + (size_t)(r0 + l16) * ld + kbase + half * 8;
  FragF F;
  F.q[0] = *(const f32x4*)(p);        // K +0..3
  F.q[1] = *(const f32x4*)(p + 4);    // K +4..7
  F.q[2] = *(const f32x4*)(p + 16);   // K +16..19
  F.q[3] = *(const f32x4*)(p + 20);   // K +20..23
  return __builtin_convertvector(F.f, v16bf);
}

// B fragment: 16-bit B 32x16 layout. lanes 0-15 K 0..7 & 16..23, lanes 16-31 K 8..15 & 24..31,
// column N = lane%16. Source: n-major bf16 (global), ld = K-extent of row.
__device__ __forceinline__ v16bf load_b(const unsigned short* m, int n0, int kbase,
                                        int ld, int lane) {
  int l16 = lane & 15, half = lane >> 4;
  const unsigned short* p = m + (size_t)(n0 + l16) * ld + kbase + half * 8;
  Frag f;
  f.u[0] = *(const u32x4*)(p);        // K +0..7
  f.u[1] = *(const u32x4*)(p + 16);   // K +16..23
  return f.v;
}

// ---------------- precompute kernels (tiny, << 1% of runtime) ----------------

// xp[b,a,ch] = block-mean of x over 14x14 pixels
__global__ void pool_kernel(const float* __restrict__ x, float* __restrict__ xp) {
  int ba = blockIdx.x, b = ba >> 4, a = ba & 15;
  int py = a >> 2, px = a & 3, ch = threadIdx.x;
  const float* xb = x + ((size_t)b * NPIX) * CDIM + ch;
  float s = 0.f;
  for (int i = 0; i < POOLB; ++i)
    for (int j = 0; j < POOLB; ++j)
      s += xb[(size_t)((py * POOLB + i) * 56 + px * POOLB + j) * CDIM];
  xp[(size_t)ba * CDIM + ch] = s * (1.f / 196.f);
}

// agent_q = xp @ Wq^T ; agent_k = xp @ Wkv^T
__global__ void agent_qk_kernel(const float* __restrict__ xp,
                                const float* __restrict__ Wq,
                                const float* __restrict__ Wkv,
                                float* __restrict__ agq, float* __restrict__ agk) {
  int ba = blockIdx.x, ch = threadIdx.x;
  const float* xr = xp + (size_t)ba * CDIM;
  const float* wq = Wq + (size_t)ch * CDIM;
  const float* wk = Wkv + (size_t)ch * CDIM;
  float aq = 0.f, ak = 0.f;
  for (int k = 0; k < CDIM; ++k) { float v = xr[k]; aq += v * wq[k]; ak += v * wk[k]; }
  agq[(size_t)ba * CDIM + ch] = aq;
  agk[(size_t)ba * CDIM + ch] = ak;
}

// A = (agent_q @ w_g) * scale ; G[b,ch] = sum_a A[a] * agent_q[b,a,ch]
__global__ void gate_kernel(const float* __restrict__ agq,
                            const float* __restrict__ w_g, float* __restrict__ G) {
  __shared__ float A[ANUM];
  int b = blockIdx.x, t = threadIdx.x;
  if (t < ANUM) {
    const float* r = agq + (size_t)(b * ANUM + t) * CDIM;
    float s = 0.f;
    for (int k = 0; k < CDIM; ++k) s += r[k] * w_g[k];
    A[t] = s * SCALE;
  }
  __syncthreads();
  float g = 0.f;
  for (int a = 0; a < ANUM; ++a) g += A[a] * agq[(size_t)(b * ANUM + a) * CDIM + t];
  G[(size_t)b * CDIM + t] = g;
}

// agent_new = (G .* agent_k) @ Wp^T + bp + agent_q
__global__ void agent_new_kernel(const float* __restrict__ G,
                                 const float* __restrict__ agk,
                                 const float* __restrict__ agq,
                                 const float* __restrict__ Wp,
                                 const float* __restrict__ bp,
                                 float* __restrict__ ann) {
  int ba = blockIdx.x, b = ba >> 4, ch = threadIdx.x;
  const float* g = G + (size_t)b * CDIM;
  const float* a = agk + (size_t)ba * CDIM;
  const float* w = Wp + (size_t)ch * CDIM;
  float s = 0.f;
  for (int k = 0; k < CDIM; ++k) s += g[k] * a[k] * w[k];
  ann[(size_t)ba * CDIM + ch] = s + bp[ch] + agq[(size_t)ba * CDIM + ch];
}

// Mt[b][h*16+a][k] = scale * sum_t Wq[h*48+t, k] * agent_new[b,a,h*48+t]  (bf16 n-major)
__global__ void build_M_kernel(const float* __restrict__ Wq,
                               const float* __restrict__ ann,
                               unsigned short* __restrict__ Mt) {
  int bn = blockIdx.x, b = bn >> 7, nn = bn & 127;
  int h = nn >> 4, a = nn & 15, k = threadIdx.x;
  const float* ar = ann + (size_t)(b * ANUM + a) * CDIM + h * DHEAD;
  const float* wq = Wq + (size_t)(h * DHEAD) * CDIM + k;
  float s = 0.f;
  for (int t = 0; t < DHEAD; ++t) s += wq[(size_t)t * CDIM] * ar[t];
  Mt[(size_t)(b * 128 + nn) * CDIM + k] = f2bf(s * SCALE);
}

// Bpt[b][j][h*16+a] = sum_t agent_k[b,a,h*48+t] * Wp[j, h*48+t]  (bf16 n-major)
__global__ void build_Bp_kernel(const float* __restrict__ agk,
                                const float* __restrict__ Wp,
                                unsigned short* __restrict__ Bpt) {
  int bj = blockIdx.x, b = bj / CDIM, j = bj % CDIM;
  int kk = threadIdx.x, h = kk >> 4, a = kk & 15;
  const float* ak = agk + (size_t)(b * ANUM + a) * CDIM + h * DHEAD;
  const float* wp = Wp + (size_t)j * CDIM + h * DHEAD;
  float s = 0.f;
  for (int t = 0; t < DHEAD; ++t) s += ak[t] * wp[t];
  Bpt[(size_t)(b * CDIM + j) * 128 + kk] = f2bf(s);
}

// ---------------- fused main kernel: L = x@M, softmax, y = S@Bp + bp --------
__global__ __launch_bounds__(256) void main_attn_kernel(
    const float* __restrict__ x, const unsigned short* __restrict__ Mt,
    const unsigned short* __restrict__ Bpt, const float* __restrict__ bp,
    float* __restrict__ out) {
  __shared__ __align__(16) float          xs[TILE_ELEMS];    // 96 KB raw f32 tile (TDM dest)
  __shared__ __align__(16) unsigned short Ss[ROWTILE * 128]; // 16 KB softmax S (bf16)

  const int wgid = blockIdx.x;
  const int b = wgid / NTILES, tile = wgid % NTILES;
  const int row0 = tile * ROWTILE;
  const int t = threadIdx.x;
  const int wave = t >> 5, lane = t & 31;
  const int half = lane >> 4, l16 = lane & 15;
  const int rt = wave & 3;        // row 16-tile within 64 rows
  const int cg = wave >> 2;       // column group (0,1)

  const unsigned short* Mb  = Mt  + (size_t)b * 128 * CDIM;
  const unsigned short* Bpb = Bpt + (size_t)b * CDIM * 128;
  const float* xg = x + ((size_t)b * NPIX + row0) * CDIM;   // contiguous 96 KB

  // ---- Tensor Data Mover: DMA the whole x tile into LDS (wave 0 issues) ----
  if (t < 32) {
    unsigned lds_off = (unsigned)(size_t)(&xs[0]);          // low 32b of flat = LDS offset
    unsigned long long ga = (unsigned long long)(size_t)xg; // 57-bit global byte address
    u32x4 g0;
    g0[0] = 1u;                                             // count=1, no gather
    g0[1] = lds_off;                                        // lds_addr
    g0[2] = (unsigned)(ga & 0xffffffffull);                 // global_addr[31:0]
    g0[3] = (unsigned)((ga >> 32) & 0x1ffffffull)           // global_addr[56:32]
            | (2u << 30);                                   // type=2 (image)
    i32x8 g1;
    g1[0] = 0x20000;                 // workgroup_mask=0, data_size=2 (4 bytes)
    g1[1] = (int)(TILE_ELEMS << 16); // tensor_dim0[15:0]=24576 in bits 63:48
    g1[2] = 0x00010000;              // tensor_dim0[31:16]=0 | tensor_dim1=1
    g1[3] = (int)(TILE_ELEMS << 16); // tile_dim0=24576 (bits 127:112)
    g1[4] = 0x00000001;              // tile_dim1=1, tile_dim2=0
    g1[5] = (int)TILE_ELEMS;         // tensor_dim0_stride[31:0]
    g1[6] = 0;
    g1[7] = 0;
    i32x4 g2 = {0, 0, 0, 0};         // 2D tensor: groups 2/3 unused
    i32x4 g3 = {0, 0, 0, 0};
#if defined(__clang_major__) && (__clang_major__ >= 23)
    i32x8 gz = {0, 0, 0, 0, 0, 0, 0, 0};
    __builtin_amdgcn_tensor_load_to_lds(g0, g1, g2, g3, gz, 0);
#else
    __builtin_amdgcn_tensor_load_to_lds(g0, g1, g2, g3, 0);
#endif
    __builtin_amdgcn_s_wait_tensorcnt(0);
  }
  // warm L2 with this batch's value matrix while the DMA runs
  __builtin_prefetch(Bpb + (size_t)t * 192, 0, 1);
  __syncthreads();

  // ---- GEMM1: L(64x128) = x(64x384) @ M(384x128), scale pre-folded ----
  v8f acc[4];
  #pragma unroll
  for (int j = 0; j < 4; ++j) acc[j] = 0.f;
  #pragma unroll 2
  for (int kk = 0; kk < CDIM / 32; ++kk) {
    v16bf A = load_a_f32(xs, rt * 16, kk * 32, CDIM, lane);
    #pragma unroll
    for (int j = 0; j < 4; ++j) {
      v16bf B = load_b(Mb, (cg * 4 + j) * 16, kk * 32, CDIM, lane);
      acc[j] = __builtin_amdgcn_wmma_f32_16x16x32_bf16(
          false, A, false, B, (short)0, acc[j], false, false);
    }
  }

  // ---- softmax over each head's 16 columns (one 16x16 C tile per head) ----
  // C layout: row = rt*16 + r + 8*half, col = ct*16 + l16.
  // Row-wise reduce across 16 lanes via ds_swizzle xor patterns (stay in half).
  #pragma unroll
  for (int j = 0; j < 4; ++j) {
    int ct = cg * 4 + j;
    #pragma unroll
    for (int r = 0; r < 8; ++r) {
      float v = acc[j][r];
      float m = v;
      m = fmaxf(m, SWZ_F(m, 0x041F));   // xor 1
      m = fmaxf(m, SWZ_F(m, 0x081F));   // xor 2
      m = fmaxf(m, SWZ_F(m, 0x101F));   // xor 4
      m = fmaxf(m, SWZ_F(m, 0x201F));   // xor 8
      float e = __expf(v - m);
      float s = e;
      s += SWZ_F(s, 0x041F);
      s += SWZ_F(s, 0x081F);
      s += SWZ_F(s, 0x101F);
      s += SWZ_F(s, 0x201F);
      float o = e / s;
      int row = rt * 16 + r + 8 * half;
      Ss[row * 128 + ct * 16 + l16] = f2bf(o);
    }
  }
  __syncthreads();

  // ---- GEMM2: y(64x384) = S(64x128) @ Bp(128x384) + bp ----
  #pragma unroll
  for (int g = 0; g < 2; ++g) {
    v8f acc2[6];
    #pragma unroll
    for (int j = 0; j < 6; ++j) acc2[j] = 0.f;
    #pragma unroll
    for (int kk = 0; kk < 4; ++kk) {
      v16bf A = load_a_bf(Ss, rt * 16, kk * 32, 128, lane);
      #pragma unroll
      for (int j = 0; j < 6; ++j) {
        int ct = cg * 12 + g * 6 + j;
        v16bf B = load_b(Bpb, ct * 16, kk * 32, 128, lane);
        acc2[j] = __builtin_amdgcn_wmma_f32_16x16x32_bf16(
            false, A, false, B, (short)0, acc2[j], false, false);
      }
    }
    #pragma unroll
    for (int j = 0; j < 6; ++j) {
      int ct = cg * 12 + g * 6 + j;
      int col = ct * 16 + l16;
      float bias = bp[col];
      float* orow = out + ((size_t)b * NPIX + row0 + rt * 16 + 8 * half) * CDIM + col;
      #pragma unroll
      for (int r = 0; r < 8; ++r)
        __builtin_nontemporal_store(acc2[j][r] + bias, &orow[(size_t)r * CDIM]);
    }
  }
}

// --------------------------------- launcher ---------------------------------
extern "C" void kernel_launch(void* const* d_in, const int* in_sizes, int n_in,
                              void* d_out, int out_size, void* d_ws, size_t ws_size,
                              hipStream_t stream) {
  const float* x   = (const float*)d_in[0];
  // d_in[1]=H, d_in[2]=W (fixed at 56)
  const float* Wq  = (const float*)d_in[3];
  const float* Wkv = (const float*)d_in[4];
  const float* w_g = (const float*)d_in[5];
  const float* Wp  = (const float*)d_in[6];
  const float* bp  = (const float*)d_in[7];
  float* out = (float*)d_out;

  char* ws = (char*)d_ws;
  float*          xp  = (float*)(ws);                       // 786432 B
  float*          agq = (float*)(ws + 786432);              // 786432 B
  float*          agk = (float*)(ws + 1572864);             // 786432 B
  float*          G   = (float*)(ws + 2359296);             // 49152 B
  float*          ann = (float*)(ws + 2408448);             // 786432 B
  unsigned short* Mt  = (unsigned short*)(ws + 3194880);    // 3145728 B
  unsigned short* Bpt = (unsigned short*)(ws + 6340608);    // 3145728 B

  pool_kernel<<<BATCH * ANUM, CDIM, 0, stream>>>(x, xp);
  agent_qk_kernel<<<BATCH * ANUM, CDIM, 0, stream>>>(xp, Wq, Wkv, agq, agk);
  gate_kernel<<<BATCH, CDIM, 0, stream>>>(agq, w_g, G);
  agent_new_kernel<<<BATCH * ANUM, CDIM, 0, stream>>>(G, agk, agq, Wp, bp, ann);
  build_M_kernel<<<BATCH * 128, CDIM, 0, stream>>>(Wq, ann, Mt);
  build_Bp_kernel<<<BATCH * CDIM, 128, 0, stream>>>(agk, Wp, Bpt);
  main_attn_kernel<<<BATCH * NTILES, 256, 0, stream>>>(x, Mt, Bpt, bp, out);
}
